// GlobalAttention_40785009443072
// MI455X (gfx1250) — matrix operand order repaired and checked by reading, not compiled, not run on hardware
//
#include <hip/hip_runtime.h>
#include <hip/hip_bf16.h>
#include <stdint.h>

// GlobalAttention for MI455X (gfx1250).
// All GEMMs via v_wmma_f32_16x16x32_bf16; B-operand tiles staged into LDS by
// the Tensor Data Mover (tensor_load_to_lds, TENSORcnt), double-buffered.
// B=16, T=1024, S=2048, D=512. Outputs: attn_h [T,B,D] f32, align [T,B,S] f32.
// Workspace requirement: ~372 MB.

typedef __bf16 bf16;
typedef __attribute__((ext_vector_type(16))) __bf16 v16bf;
typedef __attribute__((ext_vector_type(8)))  float   v8f;
typedef __attribute__((ext_vector_type(4)))  unsigned int u32x4;
typedef __attribute__((ext_vector_type(8)))  int i32x8;
typedef __attribute__((ext_vector_type(4)))  int i32x4;

#define EPI_BF16      0   // out bf16 = acc
#define EPI_SIGMOID   1   // out bf16 = sigmoid(acc + bias[col])
#define EPI_F32       2   // out f32  = acc
#define EPI_TANH_F32  3   // out f32  = tanh(acc)
#define EPI_MULZ_BF16 4   // out bf16 = acc * z[row,col]

// GEMM tile geometry.
#define BN       256          // workgroup N tile (rows of Bt staged by TDM)
#define BK       64           // K step per TDM stage (128 B rows)
#define LDS_KP   72           // BK + 8 elem pad (TDM pad: +4 dwords per 32 dwords)
#define BUF_ELEM (BN * LDS_KP)

union FragU { v16bf v; uint4 u[2]; };

// A-fragment (16-bit 16x32, ISA 7.12.2): lane L -> row L&15;
// lanes 0-15 hold K=k..k+7 (v0-3) and k+16..k+23 (v4-7); lanes 16-31 shift by 8.
__device__ __forceinline__ v16bf load_frag_a(const bf16* p) {
    FragU u;
    u.u[0] = *(const uint4*)(p);
    u.u[1] = *(const uint4*)(p + 16);
    return u.v;
}

// Same fragment, f32 source (softmax probabilities), convert to bf16.
__device__ __forceinline__ v16bf load_frag_a_f32(const float* p) {
    float4 a0 = *(const float4*)(p);
    float4 a1 = *(const float4*)(p + 4);
    float4 b0 = *(const float4*)(p + 16);
    float4 b1 = *(const float4*)(p + 20);
    v16bf v;
    v[0]=(bf16)a0.x;  v[1]=(bf16)a0.y;  v[2]=(bf16)a0.z;  v[3]=(bf16)a0.w;
    v[4]=(bf16)a1.x;  v[5]=(bf16)a1.y;  v[6]=(bf16)a1.z;  v[7]=(bf16)a1.w;
    v[8]=(bf16)b0.x;  v[9]=(bf16)b0.y;  v[10]=(bf16)b0.z; v[11]=(bf16)b0.w;
    v[12]=(bf16)b1.x; v[13]=(bf16)b1.y; v[14]=(bf16)b1.z; v[15]=(bf16)b1.w;
    return v;
}

// B-fragment (16-bit 32x16): lane L -> col L&15; lanes 0-15 hold K=k..k+7
// and k+16..k+23; lanes 16-31 shift by 8 (same order as A). Two 16 B reads (from LDS).
__device__ __forceinline__ v16bf load_frag_b(const bf16* p) {
    FragU u;
    u.u[0] = *(const uint4*)(p);
    u.u[1] = *(const uint4*)(p + 16);
    return u.v;
}

// TDM: stage a 2D bf16 tile [rows x BK] (row stride ld elems) into LDS with
// +4-dword padding every 32 dwords (LDS row stride = LDS_KP elems).
// D# layout per CDNA5 ISA 8.3/8.4 (group0: flags/lds/global addr, group1: dims).
__device__ __forceinline__ void tdm_stage_tile(unsigned lds_byte_addr,
                                               const bf16* gptr,
                                               unsigned rows, unsigned ld) {
    const unsigned long long ga = (unsigned long long)(uintptr_t)gptr;
    u32x4 g0;
    g0[0] = 1u;                                          // count=1 (valid), user mode
    g0[1] = lds_byte_addr;                               // lds_addr
    g0[2] = (unsigned)(ga & 0xFFFFFFFFu);                // global_addr[31:0]
    g0[3] = (unsigned)((ga >> 32) & 0x01FFFFFFu)         // global_addr[56:32]
          | (2u << 30);                                  // type = 2 (image)
    i32x8 g1;
    g1[0] = (int)((1u << 16)                             // data_size = 1 (2 bytes)
          | (1u << 20)                                   // pad_enable
          | (4u << 22)                                   // pad_interval: 32 dwords
          | (3u << 25));                                 // pad_amount:  4 dwords
    g1[1] = (int)((BK & 0xFFFFu) << 16);                 // tensor_dim0[15:0] = BK
    g1[2] = (int)((rows & 0xFFFFu) << 16);               // dim0 hi=0 | tensor_dim1 lo
    g1[3] = (int)(((rows >> 16) & 0xFFFFu)               // tensor_dim1 hi
          | ((unsigned)BK << 16));                       // tile_dim0 = BK
    g1[4] = (int)(rows & 0xFFFFu);                       // tile_dim1 = rows, tile_dim2=0
    g1[5] = (int)ld;                                     // tensor_dim0_stride[31:0]
    g1[6] = 0;                                           // stride0 hi | stride1 lo
    g1[7] = 0;
    const i32x4 gz4 = {0, 0, 0, 0};                      // 2D: groups 2/3 unused
    const i32x8 gz8 = {0, 0, 0, 0, 0, 0, 0, 0};
    __builtin_amdgcn_tensor_load_to_lds(g0, g1, gz4, gz4, gz8, 0);
}

// NT GEMM: Out[b, m, n] (+epi) = sum_k A[b, m, k] * Bt[b, n, k], f32 accum.
// 8 waves: 2 (M) x 4 (N); wave tile 32x64 -> 2x4 frags. WG tile 64x256.
// Bt tiles [256 x 64] staged to LDS by TDM (double buffered); A direct global.
// Grid: x = N/256, y = M/64, z = batches. M%64==0, N%256==0, K%64==0 required.
template <bool AF32>
__global__ __launch_bounds__(256) void gemm_nt(
    const void* __restrict__ Aptr, long aBatch, int lda,
    const bf16* __restrict__ Bt,   long bBatch, int ldb,
    void* __restrict__ Out, long outBatch, int outRS,
    int K, int epi,
    const float* __restrict__ bias,
    const bf16* __restrict__ Z, long zBatch, int zRS)
{
    __shared__ bf16 Bls[2 * BUF_ELEM];                  // 2 x 36 KB

    const int lane = threadIdx.x & 31;
    const int wave = threadIdx.x >> 5;
    const int wm = wave >> 2;                           // 0..1
    const int wn = wave & 3;                            // 0..3
    const int mw = blockIdx.y * 64 + wm * 32;
    const int nblk = blockIdx.x * BN;
    const int b  = blockIdx.z;
    const int rlo = lane & 15;
    const int khi = lane >> 4;

    // Global source of this block's Bt tile rows.
    const bf16* Bb = Bt + (size_t)b * bBatch + (size_t)nblk * ldb;

    // LDS fragment pointers: local col = wn*64 + j*16 + rlo, row stride LDS_KP.
    const bf16* lb = &Bls[0] + (size_t)(wn * 64 + rlo) * LDS_KP + khi * 8;

    // A fragment pointers (direct global).
    const bf16*  pa0_h = (const bf16*)Aptr  + (size_t)b * aBatch + (size_t)(mw +      rlo) * lda + khi * 8;
    const bf16*  pa1_h = (const bf16*)Aptr  + (size_t)b * aBatch + (size_t)(mw + 16 + rlo) * lda + khi * 8;
    const float* pa0_f = (const float*)Aptr + (size_t)b * aBatch + (size_t)(mw +      rlo) * lda + khi * 8;
    const float* pa1_f = (const float*)Aptr + (size_t)b * aBatch + (size_t)(mw + 16 + rlo) * lda + khi * 8;

    const v8f zero = {0.f, 0.f, 0.f, 0.f, 0.f, 0.f, 0.f, 0.f};
    v8f acc[2][4];
#pragma unroll
    for (int i = 0; i < 2; ++i)
#pragma unroll
        for (int j = 0; j < 4; ++j) acc[i][j] = zero;

    const unsigned ldsBase = (unsigned)(uintptr_t)&Bls[0];

    // Prologue: stage first B tile.
    if (wave == 0) {
        tdm_stage_tile(ldsBase, Bb, BN, (unsigned)ldb);
    }

    int cur = 0;
    for (int k = 0; k < K; k += BK, cur ^= 1) {
        // Kick off DMA for the next tile; ensure the current tile's DMA is done.
        if (wave == 0) {
            if (k + BK < K) {
                tdm_stage_tile(ldsBase + (unsigned)((cur ^ 1) * BUF_ELEM * 2),
                               Bb + (k + BK), BN, (unsigned)ldb);
                __builtin_amdgcn_s_wait_tensorcnt((short)1);
            } else {
                __builtin_amdgcn_s_wait_tensorcnt((short)0);
            }
        }
        __syncthreads();                                // current tile visible to all

        const bf16* lcur = lb + (size_t)cur * BUF_ELEM;
#pragma unroll
        for (int kk = 0; kk < BK; kk += 32) {
            v16bf a0, a1;
            if constexpr (AF32) {
                a0 = load_frag_a_f32(pa0_f + k + kk);
                a1 = load_frag_a_f32(pa1_f + k + kk);
                __builtin_prefetch((const void*)(pa0_f + k + kk + 64), 0, 1);
            } else {
                a0 = load_frag_a(pa0_h + k + kk);
                a1 = load_frag_a(pa1_h + k + kk);
                __builtin_prefetch((const void*)(pa0_h + k + kk + 64), 0, 1);
            }
            v16bf b0 = load_frag_b(lcur +  0 * LDS_KP + kk);
            v16bf b1 = load_frag_b(lcur + 16 * LDS_KP + kk);
            v16bf b2 = load_frag_b(lcur + 32 * LDS_KP + kk);
            v16bf b3 = load_frag_b(lcur + 48 * LDS_KP + kk);

            acc[0][0] = __builtin_amdgcn_wmma_f32_16x16x32_bf16(false, a0, false, b0, (short)0, acc[0][0], false, false);
            acc[0][1] = __builtin_amdgcn_wmma_f32_16x16x32_bf16(false, a0, false, b1, (short)0, acc[0][1], false, false);
            acc[0][2] = __builtin_amdgcn_wmma_f32_16x16x32_bf16(false, a0, false, b2, (short)0, acc[0][2], false, false);
            acc[0][3] = __builtin_amdgcn_wmma_f32_16x16x32_bf16(false, a0, false, b3, (short)0, acc[0][3], false, false);
            acc[1][0] = __builtin_amdgcn_wmma_f32_16x16x32_bf16(false, a1, false, b0, (short)0, acc[1][0], false, false);
            acc[1][1] = __builtin_amdgcn_wmma_f32_16x16x32_bf16(false, a1, false, b1, (short)0, acc[1][1], false, false);
            acc[1][2] = __builtin_amdgcn_wmma_f32_16x16x32_bf16(false, a1, false, b2, (short)0, acc[1][2], false, false);
            acc[1][3] = __builtin_amdgcn_wmma_f32_16x16x32_bf16(false, a1, false, b3, (short)0, acc[1][3], false, false);
        }
        __syncthreads();                                // done reading 'cur' buffer
    }

    // C/D layout (ISA 7.12.2): lane L, vgpr r -> row = r + 8*(L>>4), col = L&15.
    const int colb = nblk + wn * 64 + rlo;
    const int rowb = mw + 8 * khi;
#pragma unroll
    for (int i = 0; i < 2; ++i) {
#pragma unroll
        for (int j = 0; j < 4; ++j) {
            const int col = colb + j * 16;
#pragma unroll
            for (int r = 0; r < 8; ++r) {
                const int row = rowb + i * 16 + r;
                float v = acc[i][j][r];
                const size_t o = (size_t)b * outBatch + (size_t)row * outRS + col;
                if (epi == EPI_F32) {
                    ((float*)Out)[o] = v;
                } else if (epi == EPI_TANH_F32) {
                    ((float*)Out)[o] = tanhf(v);
                } else if (epi == EPI_BF16) {
                    ((bf16*)Out)[o] = (bf16)v;
                } else if (epi == EPI_SIGMOID) {
                    v += bias[col];
                    ((bf16*)Out)[o] = (bf16)(1.0f / (1.0f + __expf(-v)));
                } else { // EPI_MULZ_BF16
                    const float zv = (float)Z[(size_t)b * zBatch + (size_t)row * zRS + col];
                    ((bf16*)Out)[o] = (bf16)(v * zv);
                }
            }
        }
    }
}

// f32 -> bf16 conversion with independent src/dst leading dims.
__global__ __launch_bounds__(256) void cvt_bf16_kernel(
    const float* __restrict__ src, bf16* __restrict__ dst,
    long rows, int cols, int sld, int dld)
{
    const long total = rows * (long)cols;
    for (long i = (long)blockIdx.x * blockDim.x + threadIdx.x; i < total;
         i += (long)gridDim.x * blockDim.x) {
        const long r = i / cols;
        const int  c = (int)(i - r * cols);
        dst[r * (long)dld + c] = (bf16)src[r * (long)sld + c];
    }
}

// f32 [b][S][D] -> bf16 [b][D][S] transpose via LDS (32x32 tiles, 256 threads).
__global__ __launch_bounds__(256) void transpose_bf16_kernel(
    const float* __restrict__ src, bf16* __restrict__ dst, int S, int D)
{
    __shared__ float tile[32][33];
    const int b  = blockIdx.z;
    const int s0 = blockIdx.x * 32;
    const int d0 = blockIdx.y * 32;
    const int tx = threadIdx.x & 31;
    const int ty = threadIdx.x >> 5;      // 0..7
    const float* sp = src + (size_t)b * S * D;
    bf16*        dp = dst + (size_t)b * D * S;
#pragma unroll
    for (int i = ty; i < 32; i += 8)
        tile[i][tx] = sp[(size_t)(s0 + i) * D + d0 + tx];
    __syncthreads();
#pragma unroll
    for (int i = ty; i < 32; i += 8)
        dp[(size_t)(d0 + i) * S + s0 + tx] = (bf16)tile[tx][i];
}

// In-place softmax over rows of length 2048 (one workgroup per row).
__global__ __launch_bounds__(256) void softmax_rows_2048(float* __restrict__ buf)
{
    float* row = buf + (size_t)blockIdx.x * 2048;
    const int tid  = threadIdx.x;
    const int lane = tid & 31;
    const int wv   = tid >> 5;
    __shared__ float red[8];

    float v[8];
    float m = -3.402823466e38f;
#pragma unroll
    for (int i = 0; i < 8; ++i) { v[i] = row[tid + i * 256]; m = fmaxf(m, v[i]); }
#pragma unroll
    for (int off = 16; off; off >>= 1) m = fmaxf(m, __shfl_xor(m, off, 32));
    if (lane == 0) red[wv] = m;
    __syncthreads();
    m = red[0];
#pragma unroll
    for (int i = 1; i < 8; ++i) m = fmaxf(m, red[i]);
    __syncthreads();

    float s = 0.f;
#pragma unroll
    for (int i = 0; i < 8; ++i) { v[i] = __expf(v[i] - m); s += v[i]; }
#pragma unroll
    for (int off = 16; off; off >>= 1) s += __shfl_xor(s, off, 32);
    if (lane == 0) red[wv] = s;
    __syncthreads();
    s = red[0];
#pragma unroll
    for (int i = 1; i < 8; ++i) s += red[i];
    const float inv = 1.f / s;
#pragma unroll
    for (int i = 0; i < 8; ++i) row[tid + i * 256] = v[i] * inv;
}

extern "C" void kernel_launch(void* const* d_in, const int* in_sizes, int n_in,
                              void* d_out, int out_size, void* d_ws, size_t ws_size,
                              hipStream_t stream) {
    (void)in_sizes; (void)n_in; (void)out_size; (void)ws_size;

    constexpr int  B = 16, T = 1024, S = 2048, D = 512;
    constexpr long M = (long)B * T;        // 16384 total query rows
    constexpr int  F = 3 * D;              // 1536

    const float* x     = (const float*)d_in[0];
    const float* ctx   = (const float*)d_in[1];
    const float* tctx  = (const float*)d_in[2];
    const float* Win   = (const float*)d_in[3];
    const float* Wsa   = (const float*)d_in[4];
    const float* Wout  = (const float*)d_in[5];
    const float* Wgate = (const float*)d_in[6];
    const float* bgate = (const float*)d_in[7];

    float* attn  = (float*)d_out;                    // [T,B,D]
    float* align = attn + (size_t)T * B * D;         // [T,B,S]

    // Workspace carve-up (256B aligned chunks).
    char* w = (char*)d_ws;
    auto alloc = [&](size_t bytes) -> void* {
        void* p = (void*)w;
        w += (bytes + 255) & ~(size_t)255;
        return p;
    };
    bf16*  cat   = (bf16*)alloc((size_t)M * F * sizeof(bf16));       // [M,1536] = [c | x | sc]
    bf16*  qb    = (bf16*)alloc((size_t)M * D * sizeof(bf16));       // q
    bf16*  qtb   = (bf16*)alloc((size_t)M * D * sizeof(bf16));       // qt
    bf16*  zb    = (bf16*)alloc((size_t)M * D * sizeof(bf16));       // sigmoid gate
    bf16*  ctxb  = (bf16*)alloc((size_t)B * S * D * sizeof(bf16));   // [b][s][d]
    bf16*  tctxb = (bf16*)alloc((size_t)B * S * D * sizeof(bf16));
    bf16*  ctxT  = (bf16*)alloc((size_t)B * D * S * sizeof(bf16));   // [b][d][s]
    bf16*  tctxT = (bf16*)alloc((size_t)B * D * S * sizeof(bf16));
    bf16*  wib   = (bf16*)alloc((size_t)D * D * sizeof(bf16));
    bf16*  wsab  = (bf16*)alloc((size_t)D * D * sizeof(bf16));
    bf16*  wgb   = (bf16*)alloc((size_t)D * D * sizeof(bf16));
    bf16*  wob   = (bf16*)alloc((size_t)D * F * sizeof(bf16));
    float* ta    = (float*)alloc((size_t)B * T * S * sizeof(float)); // tree scores [b][t][s]

    const dim3 blk(256);
    auto cgrid = [](long total) { return dim3((unsigned)((total + 2047) / 2048)); };

    // --- Stage 1: bf16 staging ---------------------------------------------
    cvt_bf16_kernel<<<cgrid(M * D), blk, 0, stream>>>(x, cat + D, M, D, D, F); // x -> cat[:,512:1024]
    cvt_bf16_kernel<<<cgrid((long)B * S * D), blk, 0, stream>>>(ctx,  ctxb,  (long)B * S, D, D, D);
    cvt_bf16_kernel<<<cgrid((long)B * S * D), blk, 0, stream>>>(tctx, tctxb, (long)B * S, D, D, D);
    cvt_bf16_kernel<<<cgrid((long)D * D), blk, 0, stream>>>(Win,   wib,  D, D, D, D);
    cvt_bf16_kernel<<<cgrid((long)D * D), blk, 0, stream>>>(Wsa,   wsab, D, D, D, D);
    cvt_bf16_kernel<<<cgrid((long)D * D), blk, 0, stream>>>(Wgate, wgb,  D, D, D, D);
    cvt_bf16_kernel<<<cgrid((long)D * F), blk, 0, stream>>>(Wout,  wob,  D, F, F, F);
    transpose_bf16_kernel<<<dim3(S / 32, D / 32, B), blk, 0, stream>>>(ctx,  ctxT,  S, D);
    transpose_bf16_kernel<<<dim3(S / 32, D / 32, B), blk, 0, stream>>>(tctx, tctxT, S, D);

    // --- Stage 2: projections q, qt, z  (M=16384, N=512, K=512) ------------
    const dim3 gp(D / BN, (unsigned)(M / 64), 1);
    gemm_nt<false><<<gp, blk, 0, stream>>>(cat + D, 0, F, wib, 0, D,
                                           qb, 0, D, D, EPI_BF16, nullptr, nullptr, 0, 0);
    gemm_nt<false><<<gp, blk, 0, stream>>>(cat + D, 0, F, wsab, 0, D,
                                           qtb, 0, D, D, EPI_BF16, nullptr, nullptr, 0, 0);
    gemm_nt<false><<<gp, blk, 0, stream>>>(cat + D, 0, F, wgb, 0, D,
                                           zb, 0, D, D, EPI_SIGMOID, bgate, nullptr, 0, 0);

    // --- Stage 3: attention scores (per batch M=1024, N=2048, K=512) -------
    const dim3 gs(S / BN, T / 64, B);
    // semantic scores straight into d_out's align region in [T,B,S] layout
    gemm_nt<false><<<gs, blk, 0, stream>>>(qb, (long)T * D, D, ctxb, (long)S * D, D,
                                           align, (long)S, B * S, D, EPI_F32,
                                           nullptr, nullptr, 0, 0);
    gemm_nt<false><<<gs, blk, 0, stream>>>(qtb, (long)T * D, D, tctxb, (long)S * D, D,
                                           ta, (long)T * S, S, D, EPI_F32,
                                           nullptr, nullptr, 0, 0);

    // --- Stage 4: softmax in place ------------------------------------------
    softmax_rows_2048<<<dim3((unsigned)M), blk, 0, stream>>>(align);
    softmax_rows_2048<<<dim3((unsigned)M), blk, 0, stream>>>(ta);

    // --- Stage 5: weighted context (per batch M=1024, N=512, K=2048) -------
    const dim3 gc(D / BN, T / 64, B);
    // c = a @ ctx -> cat[:,0:512]
    gemm_nt<true><<<gc, blk, 0, stream>>>(align, (long)S, B * S, ctxT, (long)D * S, S,
                                          cat, (long)T * F, F, S, EPI_BF16,
                                          nullptr, nullptr, 0, 0);
    // sc = (ta @ tctx) * z -> cat[:,1024:1536]
    gemm_nt<true><<<gc, blk, 0, stream>>>(ta, (long)T * S, S, tctxT, (long)D * S, S,
                                          cat + 2 * D, (long)T * F, F, S, EPI_MULZ_BF16,
                                          nullptr, zb, (long)T * D, D);

    // --- Stage 6: output (per batch M=1024, N=512, K=1536) -----------------
    const dim3 gf(D / BN, T / 64, B);
    gemm_nt<false><<<gf, blk, 0, stream>>>(cat, (long)T * F, F, wob, 0, F,
                                           attn, (long)D, B * D, F, EPI_TANH_F32,
                                           nullptr, nullptr, 0, 0);
}
